// GRU_34041910788215
// MI455X (gfx1250) — matrix operand order repaired and checked by reading, not compiled, not run on hardware
//
#include <hip/hip_runtime.h>
#include <hip/hip_bf16.h>
#include <math.h>

// GRU on MI455X (gfx1250, wave32, WMMA).
// Pipeline: prep (weights -> f16 transposed) -> GEMM1 (x -> gate/cand pre-acts, WMMA,
//           output re-laid-out [t][32][768]) -> persistent scan (2 WGs x 24 waves,
//           half register-resident / half L2-streamed recurrent weights,
//           v_wmma_f32_16x16x32_f16 per step) -> projection GEMM (async LDS stage) + sigmoid.
// Workspace layout (needs ~139 MB):
//   [0x000000) WxcT  f16 [768][256]     (input weights, transposed, gates||cand)
//   [0x060000) WhT   f16 [768][256]     (recurrent weights, transposed, gates||cand)
//   [0x0C0000) WpT   f16 [128][256]     (projection weights, transposed)
//   [0x0D0000) biasxc f32 [768]         (b_gate || b_cand)
//   [0x400000) gxc   f16 [2048][32][768] (input pre-activations, timestep-major)
//   [0x400000+96MB) ys f16 [65536][256]  (hidden states, row = b*T+t)

#define B_  32
#define T_  2048
#define D_  256
#define H_  256
#define C_  128

typedef __attribute__((ext_vector_type(16))) _Float16 v16h;
typedef __attribute__((ext_vector_type(8)))  float    v8f;
typedef __attribute__((ext_vector_type(2)))  _Float16 h2v;

__device__ __forceinline__ v8f wmma16(v16h a, v16h b, v8f c) {
  // D = A(16x32 f16) * B(32x16 f16) + C(16x16 f32)
  return __builtin_amdgcn_wmma_f32_16x16x32_f16(false, a, false, b, (short)0, c,
                                                false, false);
}

// A-fragment (16x32 f16) from row-major src (ld halfs).
// Layout (ISA 7.12.2): lane L holds row M = rowBase + L%16;
// VGPR v<4 -> K = 2v + 8*(L/16); v>=4 -> K = 16 + 2(v-4) + 8*(L/16).
__device__ __forceinline__ v16h load_A(const _Float16* src, int ld, int rowBase,
                                       int kBase, int lane) {
  const int m  = rowBase + (lane & 15);
  const int hh = (lane >> 4) * 8;
  const h2v* p0 = (const h2v*)(src + (size_t)m * ld + kBase + hh);
  const h2v* p1 = (const h2v*)(src + (size_t)m * ld + kBase + 16 + hh);
  v16h a;
#pragma unroll
  for (int v = 0; v < 4; ++v) { h2v t = p0[v]; a[2*v]   = t[0]; a[2*v+1]   = t[1]; }
#pragma unroll
  for (int v = 0; v < 4; ++v) { h2v t = p1[v]; a[8+2*v] = t[0]; a[8+2*v+1] = t[1]; }
  return a;
}

// B-fragment (32x16 f16) from transposed weights BT row-major [N][K] (ld halfs).
// Layout: lane L holds column N (nCol already includes L%16);
// VGPR v<4 -> K = 2v + 8*(L/16); v>=4 -> K = 16 + 2(v-4) + 8*(L/16).
__device__ __forceinline__ v16h load_B(const _Float16* bt, int ld, int nCol,
                                       int kBase, int lane) {
  const h2v* p = (const h2v*)(bt + (size_t)nCol * ld + kBase + (lane >> 4) * 8);
  v16h b;
#pragma unroll
  for (int v = 0; v < 8; ++v) { h2v t = p[(v & 3) + (v >> 2) * 8]; b[2*v] = t[0]; b[2*v+1] = t[1]; }
  return b;
}

// Fast activations: v_exp_f32 + v_rcp_f32, exact in the +-inf limits.
__device__ __forceinline__ float fast_sigmoid(float x) {
  return __builtin_amdgcn_rcpf(1.0f + __expf(-x));
}
__device__ __forceinline__ float fast_tanh(float x) {
  return 1.0f - 2.0f * __builtin_amdgcn_rcpf(1.0f + __expf(2.0f * x));
}

// ---------------- weight prep: f32 -> f16, transpose, concatenate ----------------
__global__ void prep_kernel(const float* __restrict__ W_gate,
                            const float* __restrict__ b_gate,
                            const float* __restrict__ W_cand,
                            const float* __restrict__ b_cand,
                            const float* __restrict__ W_proj,
                            _Float16* __restrict__ WxcT,
                            _Float16* __restrict__ WhT,
                            _Float16* __restrict__ WpT,
                            float* __restrict__ biasxc) {
  const int stride = gridDim.x * blockDim.x;
  const int i0 = blockIdx.x * blockDim.x + threadIdx.x;
  // WxcT/WhT rows 0..511 = gates (from W_gate), rows 512..767 = cand (from W_cand)
  for (int i = i0; i < 768 * 256; i += stride) {
    const int n = i >> 8, k = i & 255;
    float vx, vh;
    if (n < 512) { vx = W_gate[k * 512 + n];         vh = W_gate[(256 + k) * 512 + n]; }
    else         { vx = W_cand[k * 256 + (n - 512)]; vh = W_cand[(256 + k) * 256 + (n - 512)]; }
    WxcT[i] = (_Float16)vx;
    WhT[i]  = (_Float16)vh;
  }
  for (int i = i0; i < 128 * 256; i += stride) {
    const int n = i >> 8, k = i & 255;
    WpT[i] = (_Float16)W_proj[k * 128 + n];
  }
  for (int i = i0; i < 768; i += stride)
    biasxc[i] = (i < 512) ? b_gate[i] : b_cand[i - 512];
}

// ---------------- generic 32-row-block WMMA GEMM (K=256) ----------------
// C = act( A[M x 256] * BT^T + bias ). A staged to LDS (f32 path converts;
// f16 path uses CDNA5 async global->LDS DMA). 8 waves, 2 M-tiles each,
// N/16/8 N-tiles each, 8 K-slices of v_wmma_f32_16x16x32_f16.
// permRows: output row (b*T+t) stored at (t*32 + b) instead (timestep-major).
__global__ __launch_bounds__(256, 1) void gemm_rowblk32(
    const float* __restrict__ A32, const _Float16* __restrict__ A16,
    const _Float16* __restrict__ BT, const float* __restrict__ bias,
    _Float16* __restrict__ C16, float* __restrict__ C32, int N, int act,
    int permRows) {
  __shared__ _Float16 xs[32 * 256];
  const int tid = threadIdx.x;
  const int rowBase = blockIdx.x * 32;
  if (A32) {
    for (int i = tid; i < 32 * 256; i += 256)
      xs[i] = (_Float16)A32[(size_t)rowBase * 256 + i];
  } else {
    // CDNA5 async DMA: 16 KB global -> LDS, ASYNCcnt-tracked, 16 B per issue.
    const char* gsrc = (const char*)(A16 + (size_t)rowBase * 256);
    const unsigned ldsBase = (unsigned)(size_t)(&xs[0]);  // low 32 bits = LDS byte addr
#pragma unroll
    for (int it = 0; it < 4; ++it) {
      const int off = tid * 16 + it * 4096;
      const unsigned ldsAddr = ldsBase + off;
      const char* g = gsrc + off;
      asm volatile("global_load_async_to_lds_b128 %0, %1, off"
                   :: "v"(ldsAddr), "v"(g) : "memory");
    }
    asm volatile("s_wait_asynccnt 0x0" ::: "memory");
  }
  __syncthreads();

  const int wave = tid >> 5, lane = tid & 31;
  const int nTiles = N >> 4;
  for (int nt = wave; nt < nTiles; nt += 8) {
    const int nCol = nt * 16 + (lane & 15);
    const float bv = bias[nCol];
    v8f acc0, acc1;
#pragma unroll
    for (int v = 0; v < 8; ++v) { acc0[v] = bv; acc1[v] = bv; }
#pragma unroll
    for (int s = 0; s < 8; ++s) {
      v16h a0 = load_A(xs, 256, 0,  32 * s, lane);
      v16h a1 = load_A(xs, 256, 16, 32 * s, lane);
      v16h b  = load_B(BT, 256, nCol, 32 * s, lane);
      acc0 = wmma16(a0, b, acc0);
      acc1 = wmma16(a1, b, acc1);
    }
#pragma unroll
    for (int v = 0; v < 8; ++v) {
      const int m = v + ((lane >> 4) << 3);
      float x0 = acc0[v], x1 = acc1[v];
      if (act) { x0 = fast_sigmoid(x0); x1 = fast_sigmoid(x1); }
      const int rg0 = rowBase + m, rg1 = rowBase + 16 + m;
      const size_t r0 = permRows ? ((size_t)(rg0 & (T_ - 1)) * 32 + (rg0 >> 11))
                                 : (size_t)rg0;
      const size_t r1 = permRows ? ((size_t)(rg1 & (T_ - 1)) * 32 + (rg1 >> 11))
                                 : (size_t)rg1;
      if (C16) { C16[r0 * N + nCol] = (_Float16)x0; C16[r1 * N + nCol] = (_Float16)x1; }
      else     { C32[r0 * N + nCol] = x0;           C32[r1 * N + nCol] = x1; }
    }
  }
}

// ---------------- persistent recurrent scan ----------------
// 2 workgroups (16 batch rows each) x 768 threads (24 wave32).
// Waves 0-15: gate matmul (512 cols), waves 0-7 produce r (write r*h to LDS),
// waves 8-15 hold u and do the h update. Waves 16-23: candidate matmul (256 cols).
// Recurrent weights: K-slices 0-3 register-resident (64 VGPRs/wave, safe for
// 24 resident waves), slices 4-7 streamed from L2 each step (pipelines with WMMA).
// gxc is timestep-major: one contiguous 48 KB slab per step, prefetched.
__global__ __launch_bounds__(768, 1) void gru_scan(
    const _Float16* __restrict__ gxc, const _Float16* __restrict__ WhT,
    _Float16* __restrict__ ys) {
  __shared__ _Float16 h_lds [16 * 256];
  __shared__ _Float16 rh_lds[16 * 256];
  __shared__ _Float16 c_lds [16 * 256];
  const int tid = threadIdx.x, wave = tid >> 5, lane = tid & 31;
  const int bBase = blockIdx.x * 16;
  for (int i = tid; i < 16 * 256; i += 768) {
    h_lds[i]  = (_Float16)0.0f;
    rh_lds[i] = (_Float16)0.0f;
  }
  __syncthreads();

  const bool isCand   = (wave >= 16);
  const int  tb       = isCand ? (wave - 16) : wave;
  const int  gColBase = (isCand ? 512 : 0) + tb * 32;  // col in gxc / row in WhT
  const int  mRow     = (lane >> 4) << 3;              // C-layout row base (+v)
  const int  wRow0    = gColBase + (lane & 15);        // lane's WhT row, tile 0

  // Register-resident recurrent weights: 2 n-tiles x K-slices 0..3.
  v16h Breg[2][4];
#pragma unroll
  for (int nt = 0; nt < 2; ++nt)
#pragma unroll
    for (int s = 0; s < 4; ++s)
      Breg[nt][s] = load_B(WhT, 256, wRow0 + nt * 16, 32 * s, lane);

  for (int t = 0; t < T_; ++t) {
    const _Float16* slab = gxc + (size_t)t * 32 * 768;
    // init accumulators from precomputed input pre-activations (contiguous slab)
    v8f acc[2];
#pragma unroll
    for (int nt = 0; nt < 2; ++nt) {
      const int col = gColBase + nt * 16 + (lane & 15);
#pragma unroll
      for (int v = 0; v < 8; ++v)
        acc[nt][v] = (float)slab[(size_t)(bBase + mRow + v) * 768 + col];
    }
    if (t + 1 < T_)  // one 64 B line per thread covers the next 48 KB slab
      __builtin_prefetch(gxc + (size_t)(t + 1) * 32 * 768 + tid * 32, 0, 1);

    if (!isCand) {
      // gates = sigmoid(gx_t + h @ Wh_g)
#pragma unroll
      for (int s = 0; s < 8; ++s) {
        v16h a = load_A(h_lds, 256, 0, 32 * s, lane);
        v16h b0 = (s < 4) ? Breg[0][s & 3] : load_B(WhT, 256, wRow0,      32 * s, lane);
        v16h b1 = (s < 4) ? Breg[1][s & 3] : load_B(WhT, 256, wRow0 + 16, 32 * s, lane);
        acc[0] = wmma16(a, b0, acc[0]);
        acc[1] = wmma16(a, b1, acc[1]);
      }
#pragma unroll
      for (int nt = 0; nt < 2; ++nt)
#pragma unroll
        for (int v = 0; v < 8; ++v) acc[nt][v] = fast_sigmoid(acc[nt][v]);
      if (wave < 8) {  // r-waves: rh = r * h
#pragma unroll
        for (int nt = 0; nt < 2; ++nt) {
          const int col = gColBase + nt * 16 + (lane & 15);
#pragma unroll
          for (int v = 0; v < 8; ++v) {
            const int idx = (mRow + v) * 256 + col;
            rh_lds[idx] = (_Float16)(acc[nt][v] * (float)h_lds[idx]);
          }
        }
      }
    }
    __syncthreads();  // rh ready

    if (isCand) {
      // c = tanh(cx_t + (r*h) @ Wh_c)
#pragma unroll
      for (int s = 0; s < 8; ++s) {
        v16h a = load_A(rh_lds, 256, 0, 32 * s, lane);
        v16h b0 = (s < 4) ? Breg[0][s & 3] : load_B(WhT, 256, wRow0,      32 * s, lane);
        v16h b1 = (s < 4) ? Breg[1][s & 3] : load_B(WhT, 256, wRow0 + 16, 32 * s, lane);
        acc[0] = wmma16(a, b0, acc[0]);
        acc[1] = wmma16(a, b1, acc[1]);
      }
#pragma unroll
      for (int nt = 0; nt < 2; ++nt) {
        const int col = (gColBase - 512) + nt * 16 + (lane & 15);
#pragma unroll
        for (int v = 0; v < 8; ++v)
          c_lds[(mRow + v) * 256 + col] = (_Float16)fast_tanh(acc[nt][v]);
      }
    }
    __syncthreads();  // c ready

    if (!isCand && wave >= 8) {  // u-waves: h = u*h + (1-u)*c, emit ys
#pragma unroll
      for (int nt = 0; nt < 2; ++nt) {
        const int col = (gColBase - 256) + nt * 16 + (lane & 15);
#pragma unroll
        for (int v = 0; v < 8; ++v) {
          const int idx = (mRow + v) * 256 + col;
          const float u  = acc[nt][v];
          const float hn = u * (float)h_lds[idx] + (1.0f - u) * (float)c_lds[idx];
          const _Float16 hh = (_Float16)hn;
          h_lds[idx] = hh;
          ys[(size_t)((bBase + mRow + v) * T_ + t) * 256 + col] = hh;
        }
      }
    }
    __syncthreads();  // h updated
  }
}

extern "C" void kernel_launch(void* const* d_in, const int* in_sizes, int n_in,
                              void* d_out, int out_size, void* d_ws, size_t ws_size,
                              hipStream_t stream) {
  (void)in_sizes; (void)n_in; (void)out_size; (void)ws_size;
  const float* x      = (const float*)d_in[0];
  const float* W_gate = (const float*)d_in[1];
  const float* b_gate = (const float*)d_in[2];
  const float* W_cand = (const float*)d_in[3];
  const float* b_cand = (const float*)d_in[4];
  const float* W_proj = (const float*)d_in[5];
  const float* b_proj = (const float*)d_in[6];

  char* ws = (char*)d_ws;
  _Float16* WxcT   = (_Float16*)(ws + 0x000000);
  _Float16* WhT    = (_Float16*)(ws + 0x060000);
  _Float16* WpT    = (_Float16*)(ws + 0x0C0000);
  float*    biasxc = (float*)   (ws + 0x0D0000);
  _Float16* gxc    = (_Float16*)(ws + 0x400000);
  _Float16* ys     = (_Float16*)(ws + 0x400000 + (size_t)65536 * 768 * 2);

  prep_kernel<<<512, 256, 0, stream>>>(W_gate, b_gate, W_cand, b_cand, W_proj,
                                       WxcT, WhT, WpT, biasxc);
  // gxc[t][b][0:768] = x @ [WxT_g | WxT_c] + [b_gate | b_cand] (timestep-major)
  gemm_rowblk32<<<2048, 256, 0, stream>>>(x, nullptr, WxcT, biasxc,
                                          gxc, nullptr, 768, 0, 1);
  // sequential recurrence, 2 persistent workgroups (16 batch rows each)
  gru_scan<<<2, 768, 0, stream>>>(gxc, WhT, ys);
  // out = sigmoid(ys @ W_proj + b_proj)
  gemm_rowblk32<<<2048, 256, 0, stream>>>(nullptr, ys, WpT, b_proj,
                                          nullptr, (float*)d_out, 128, 1, 0);
}
